// PCGCNN_54717883351119
// MI455X (gfx1250) — matrix operand order repaired and checked by reading, not compiled, not run on hardware
//
#include <hip/hip_runtime.h>
#include <hip/hip_bf16.h>

typedef __attribute__((ext_vector_type(16))) __bf16 v16bf;
typedef __attribute__((ext_vector_type(8)))  float  v8f;

#define N_   4096
#define D_   127
#define H_   128
#define L_   2
#define OUT_ 3

__device__ __forceinline__ unsigned short f2bf(float f) {
  unsigned u = __float_as_uint(f);
  u += 0x7FFFu + ((u >> 16) & 1u);          // round-to-nearest-even
  return (unsigned short)(u >> 16);
}

union Frag {
  unsigned u[8];
  uint4 q[2];
  v16bf v;
};

// ---------------- prep: build h0 (f32 + bf16) and normalized 4-dim feature fn
__global__ void k_prep(const float* __restrict__ x, float* __restrict__ hF,
                       unsigned short* __restrict__ hB, float* __restrict__ fn) {
  int idx = blockIdx.x * 256 + threadIdx.x;
  if (idx >= N_ * H_) return;
  int i = idx >> 7, c = idx & 127;
  float v = (c == 0) ? x[i * D_] : x[i * D_ + c - 1];
  hF[idx] = v;
  hB[idx] = f2bf(v);
  if (c == 0) {
    float a = x[i * D_ + 0], b = x[i * D_ + 1], cc = x[i * D_ + 2];
    float nrm = fmaxf(sqrtf(2.f * a * a + b * b + cc * cc), 1e-12f);
    ((float4*)fn)[i] = make_float4(a / nrm, a / nrm, b / nrm, cc / nrm);
  }
}

// ---------------- convert weights to bf16
__global__ void k_wcvt(const float* __restrict__ Wl, const float* __restrict__ Wr,
                       unsigned short* __restrict__ WlB, unsigned short* __restrict__ WrB) {
  int idx = blockIdx.x * 256 + threadIdx.x;
  if (idx < L_ * H_ * H_) {
    WlB[idx] = f2bf(Wl[idx]);
    WrB[idx] = f2bf(Wr[idx]);
  }
}

// ---------------- adjacency bitmask: bit j of bits[i*128 + j/32]
__global__ void k_adj(const float* __restrict__ fn, const int* __restrict__ sat,
                      unsigned* __restrict__ bits) {
  int idx = blockIdx.x * 256 + threadIdx.x;
  if (idx >= N_ * 128) return;
  int i = idx >> 7, w = idx & 127;
  float4 fi = ((const float4*)fn)[i];
  int si = sat[i];
  unsigned word = 0;
  int jbase = w * 32;
  #pragma unroll 4
  for (int jj = 0; jj < 32; ++jj) {
    int j = jbase + jj;
    float4 fj = ((const float4*)fn)[j];
    float d = fi.x * fj.x + fi.y * fj.y + fi.z * fj.z + fi.w * fj.w;
    bool m = ((si == sat[j]) | (d > 0.9f)) && (j != i);
    word |= (m ? 1u : 0u) << jj;
  }
  bits[idx] = word;
}

// ---------------- degrees -> 1/max(deg,1)
__global__ void k_deg(const unsigned* __restrict__ bits, float* __restrict__ invdeg) {
  int i = blockIdx.x * 256 + threadIdx.x;
  if (i >= N_) return;
  int s = 0;
  #pragma unroll 8
  for (int w = 0; w < 128; ++w) s += __popc(bits[i * 128 + w]);
  float d = (s < 1) ? 1.0f : (float)s;
  invdeg[i] = 1.0f / d;
}

// ---------------- GEMM1 partial: mP[kq] = A[:,kq] @ h[kq, nhalf]
// grid (16 Mblk, 2 Nhalf, 4 Kquarter), 256 thr = 8 waves
// wave: 32 rows (2 A-frags) x 4 N-tiles (8 accums), K = 1024
// double-buffered LDS staging, 1 barrier / K-step, 2-deep global pipeline
#define PADK 40
__global__ void k_gemm1(const unsigned* __restrict__ bits,
                        const unsigned short* __restrict__ hB,
                        float* __restrict__ mP) {
  __shared__ unsigned short ldsB[2][64 * PADK];   // transposed [n][k]
  const int tid = threadIdx.x;
  const int wave = tid >> 5, lane = tid & 31;
  const int half_ = lane >> 4, mrow = lane & 15;
  const int i0 = blockIdx.x * 256 + wave * 32;
  const int n0 = blockIdx.y * 64;
  const int kbase = blockIdx.z * 1024;
  const int kend = kbase + 1024;
  float* outp = mP + (size_t)blockIdx.z * (N_ * H_);

  v8f z = {0, 0, 0, 0, 0, 0, 0, 0};
  v8f acc[2][4];
  #pragma unroll
  for (int a = 0; a < 2; ++a)
    #pragma unroll
    for (int t = 0; t < 4; ++t) acc[a][t] = z;

  // staging mapping: sr = row in K-tile (0..31), sc = 8-half chunk (0..7)
  const int sr = tid >> 3, sc = tid & 7;
  union St { uint4 q; unsigned short s[8]; };
  St cur;
  cur.q = *(const uint4*)(hB + (kbase + sr) * H_ + n0 + sc * 8);
  #pragma unroll
  for (int e = 0; e < 8; ++e)
    ldsB[0][(sc * 8 + e) * PADK + sr] = cur.s[e];
  __syncthreads();
  cur.q = *(const uint4*)(hB + (kbase + 32 + sr) * H_ + n0 + sc * 8);

  for (int kb = kbase; kb < kend; kb += 32) {
    const int p = (kb >> 5) & 1;
    if (kb + 32 < kend) {       // stage tile kb+32 into other buffer
      #pragma unroll
      for (int e = 0; e < 8; ++e)
        ldsB[p ^ 1][(sc * 8 + e) * PADK + sr] = cur.s[e];
    }
    if (kb + 64 < kend)         // keep global pipeline 2 tiles deep
      cur.q = *(const uint4*)(hB + (kb + 64 + sr) * H_ + n0 + sc * 8);

    // two A fragments (rows i0.., i0+16..) from one bitmask word per lane each
    unsigned w0 = bits[(i0 + mrow) * 128 + (kb >> 5)];
    unsigned w1 = bits[(i0 + 16 + mrow) * 128 + (kb >> 5)];
    Frag a0, a1;
    #pragma unroll
    for (int r2 = 0; r2 < 8; ++r2) {
      int kba = 2 * r2 + (half_ ? 16 : 0);
      a0.u[r2] = (((w0 >> kba) & 1u) ? 0x3F80u : 0u) |
                 (((w0 >> (kba + 1)) & 1u) ? 0x3F800000u : 0u);
      a1.u[r2] = (((w1 >> kba) & 1u) ? 0x3F80u : 0u) |
                 (((w1 >> (kba + 1)) & 1u) ? 0x3F800000u : 0u);
    }

    #pragma unroll
    for (int t = 0; t < 4; ++t) {     // each B fragment feeds 2 WMMAs
      Frag bf_;
      const unsigned short* ls = &ldsB[p][(t * 16 + mrow) * PADK + half_ * 16];
      bf_.q[0] = *(const uint4*)ls;
      bf_.q[1] = *(const uint4*)(ls + 8);
      acc[0][t] = __builtin_amdgcn_wmma_f32_16x16x32_bf16(
          false, a0.v, false, bf_.v, (short)0, acc[0][t], false, false);
      acc[1][t] = __builtin_amdgcn_wmma_f32_16x16x32_bf16(
          false, a1.v, false, bf_.v, (short)0, acc[1][t], false, false);
    }
    __syncthreads();
  }

  #pragma unroll
  for (int a = 0; a < 2; ++a)
    #pragma unroll
    for (int r = 0; r < 8; ++r) {
      int i = i0 + a * 16 + r + (half_ ? 8 : 0);
      #pragma unroll
      for (int t = 0; t < 4; ++t)
        outp[(size_t)i * H_ + n0 + t * 16 + mrow] = acc[a][t][r];
    }
}

// ---------------- combine K-split partials, scale by 1/deg, convert to bf16
__global__ void k_comb(const float* __restrict__ mP, const float* __restrict__ invdeg,
                       unsigned short* __restrict__ mB) {
  int idx = blockIdx.x * 256 + threadIdx.x;
  if (idx >= N_ * H_) return;
  float v = (mP[idx] + mP[N_ * H_ + idx] + mP[2 * N_ * H_ + idx] + mP[3 * N_ * H_ + idx])
            * invdeg[idx >> 7];
  mB[idx] = f2bf(v);
}

// ---------------- GEMM2: h_out = relu(mB@Wl^T + bl + hB_in@Wr^T)
// grid (32 Mblk, 2 Nhalf), 256 thr = 8 waves; wave: 16 rows x 4 N-tiles, K=128
#define PADW 136
__global__ void k_gemm2(const unsigned short* __restrict__ mB,
                        const unsigned short* __restrict__ hBin,
                        const unsigned short* __restrict__ WlB,
                        const unsigned short* __restrict__ WrB,
                        const float* __restrict__ bl,
                        float* __restrict__ hF,
                        unsigned short* __restrict__ hBout) {
  __shared__ unsigned short ldsW[64 * PADW];    // [o][k], rows n0..n0+63
  const int tid = threadIdx.x;
  const int wave = tid >> 5, lane = tid & 31;
  const int half_ = lane >> 4, mrow = lane & 15;
  const int i0 = blockIdx.x * 128 + wave * 16;
  const int n0 = blockIdx.y * 64;

  v8f z = {0, 0, 0, 0, 0, 0, 0, 0};
  v8f acc[4];
  #pragma unroll
  for (int t = 0; t < 4; ++t) acc[t] = z;

  #pragma unroll
  for (int ph = 0; ph < 2; ++ph) {
    const unsigned short* W = ph ? WrB : WlB;
    const unsigned short* Am = ph ? hBin : mB;
    __syncthreads();
    { // stage 64x128 bf16 weight rows [n0..n0+64) into LDS
      int r = tid >> 2, c = (tid & 3) * 32;
      const uint4* src = (const uint4*)(W + (n0 + r) * H_ + c);
      uint4* dst = (uint4*)(&ldsW[r * PADW + c]);
      #pragma unroll
      for (int e = 0; e < 4; ++e) dst[e] = src[e];
    }
    __syncthreads();
    #pragma unroll
    for (int kb = 0; kb < H_; kb += 32) {
      Frag af;
      const unsigned short* as = Am + (i0 + mrow) * H_ + kb + half_ * 16;
      af.q[0] = *(const uint4*)as;
      af.q[1] = *(const uint4*)(as + 8);
      Frag bf0, bf1, bf2, bf3;
      {
        const unsigned short* l0 = &ldsW[(0 * 16 + mrow) * PADW + kb + half_ * 16];
        const unsigned short* l1 = &ldsW[(1 * 16 + mrow) * PADW + kb + half_ * 16];
        const unsigned short* l2 = &ldsW[(2 * 16 + mrow) * PADW + kb + half_ * 16];
        const unsigned short* l3 = &ldsW[(3 * 16 + mrow) * PADW + kb + half_ * 16];
        bf0.q[0] = *(const uint4*)l0; bf0.q[1] = *(const uint4*)(l0 + 8);
        bf1.q[0] = *(const uint4*)l1; bf1.q[1] = *(const uint4*)(l1 + 8);
        bf2.q[0] = *(const uint4*)l2; bf2.q[1] = *(const uint4*)(l2 + 8);
        bf3.q[0] = *(const uint4*)l3; bf3.q[1] = *(const uint4*)(l3 + 8);
      }
      acc[0] = __builtin_amdgcn_wmma_f32_16x16x32_bf16(false, af.v, false, bf0.v,
                                                       (short)0, acc[0], false, false);
      acc[1] = __builtin_amdgcn_wmma_f32_16x16x32_bf16(false, af.v, false, bf1.v,
                                                       (short)0, acc[1], false, false);
      acc[2] = __builtin_amdgcn_wmma_f32_16x16x32_bf16(false, af.v, false, bf2.v,
                                                       (short)0, acc[2], false, false);
      acc[3] = __builtin_amdgcn_wmma_f32_16x16x32_bf16(false, af.v, false, bf3.v,
                                                       (short)0, acc[3], false, false);
    }
  }

  #pragma unroll
  for (int t = 0; t < 4; ++t) {
    int col = n0 + t * 16 + mrow;
    float bias = bl[col];
    #pragma unroll
    for (int r = 0; r < 8; ++r) {
      int i = i0 + r + (half_ ? 8 : 0);
      float v = fmaxf(acc[t][r] + bias, 0.f);
      hF[(size_t)i * H_ + col] = v;
      hBout[(size_t)i * H_ + col] = f2bf(v);
    }
  }
}

// ---------------- batchnorm stats per column: stats[c]=mu, stats[128+c]=rsqrt(var+eps)
__global__ void k_bn(const float* __restrict__ hF, float* __restrict__ stats) {
  __shared__ float ss[256], ss2[256];
  int c = blockIdx.x;
  float s = 0.f, s2 = 0.f;
  for (int r = threadIdx.x; r < N_; r += 256) {
    float v = hF[r * H_ + c];
    s += v; s2 += v * v;
  }
  ss[threadIdx.x] = s; ss2[threadIdx.x] = s2;
  __syncthreads();
  for (int o = 128; o > 0; o >>= 1) {
    if (threadIdx.x < o) {
      ss[threadIdx.x] += ss[threadIdx.x + o];
      ss2[threadIdx.x] += ss2[threadIdx.x + o];
    }
    __syncthreads();
  }
  if (threadIdx.x == 0) {
    float mu = ss[0] / (float)N_;
    float var = ss2[0] / (float)N_ - mu * mu;
    stats[c] = mu;
    stats[H_ + c] = rsqrtf(var + 1e-5f);
  }
}

// ---------------- normalize into d_out[0 .. N*H)
__global__ void k_norm(const float* __restrict__ hF, const float* __restrict__ stats,
                       const float* __restrict__ gamma, const float* __restrict__ beta,
                       float* __restrict__ out) {
  int idx = blockIdx.x * 256 + threadIdx.x;
  if (idx >= N_ * H_) return;
  int c = idx & 127;
  out[idx] = (hF[idx] - stats[c]) * stats[H_ + c] * gamma[c] + beta[c];
}

// ---------------- final 128->3 projection into d_out[N*H ..)
__global__ void k_out(const float* __restrict__ hn, const float* __restrict__ Wo,
                      const float* __restrict__ bo, float* __restrict__ out) {
  int idx = blockIdx.x * 256 + threadIdx.x;
  if (idx >= N_ * OUT_) return;
  int i = idx / 3, o = idx - 3 * i;
  const float* hr = hn + i * H_;
  const float* wr = Wo + o * H_;
  float s = bo[o];
  #pragma unroll 8
  for (int k = 0; k < H_; ++k) s += hr[k] * wr[k];
  out[idx] = s;
}

extern "C" void kernel_launch(void* const* d_in, const int* in_sizes, int n_in,
                              void* d_out, int out_size, void* d_ws, size_t ws_size,
                              hipStream_t stream) {
  const float* x     = (const float*)d_in[0];
  const int*   sat   = (const int*)d_in[1];
  const float* Wl    = (const float*)d_in[2];
  const float* bl    = (const float*)d_in[3];
  const float* Wr    = (const float*)d_in[4];
  const float* gamma = (const float*)d_in[5];
  const float* beta  = (const float*)d_in[6];
  const float* Wo    = (const float*)d_in[7];
  const float* bo    = (const float*)d_in[8];
  float* out = (float*)d_out;

  char* ws = (char*)d_ws;
  size_t off = 0;
  auto alloc = [&](size_t bytes) {
    void* p = ws + off;
    off = (off + bytes + 255) & ~(size_t)255;
    return p;
  };
  float*          hF     = (float*)alloc((size_t)N_ * H_ * 4);
  unsigned short* hBa    = (unsigned short*)alloc((size_t)N_ * H_ * 2);
  unsigned short* hBb    = (unsigned short*)alloc((size_t)N_ * H_ * 2);
  unsigned short* mB     = (unsigned short*)alloc((size_t)N_ * H_ * 2);
  float*          mP     = (float*)alloc((size_t)4 * N_ * H_ * 4);   // K-split partials
  unsigned*       bits   = (unsigned*)alloc((size_t)N_ * 128 * 4);
  float*          fn     = (float*)alloc((size_t)N_ * 4 * 4);
  float*          invdeg = (float*)alloc((size_t)N_ * 4);
  unsigned short* WlB    = (unsigned short*)alloc((size_t)L_ * H_ * H_ * 2);
  unsigned short* WrB    = (unsigned short*)alloc((size_t)L_ * H_ * H_ * 2);
  float*          stats  = (float*)alloc((size_t)2 * H_ * 4);

  k_prep<<<(N_ * H_) / 256, 256, 0, stream>>>(x, hF, hBa, fn);
  k_wcvt<<<(L_ * H_ * H_ + 255) / 256, 256, 0, stream>>>(Wl, Wr, WlB, WrB);
  k_adj<<<(N_ * 128) / 256, 256, 0, stream>>>(fn, sat, bits);
  k_deg<<<(N_ + 255) / 256, 256, 0, stream>>>(bits, invdeg);

  unsigned short* hcur = hBa;
  unsigned short* hnxt = hBb;
  for (int l = 0; l < L_; ++l) {
    k_gemm1<<<dim3(N_ / 256, 2, 4), 256, 0, stream>>>(bits, hcur, mP);
    k_comb<<<(N_ * H_) / 256, 256, 0, stream>>>(mP, invdeg, mB);
    k_gemm2<<<dim3(N_ / 128, 2), 256, 0, stream>>>(mB, hcur, WlB + l * H_ * H_,
                                                   WrB + l * H_ * H_, bl + l * H_,
                                                   hF, hnxt);
    unsigned short* t = hcur; hcur = hnxt; hnxt = t;
  }

  k_bn<<<H_, 256, 0, stream>>>(hF, stats);
  k_norm<<<(N_ * H_) / 256, 256, 0, stream>>>(hF, stats, gamma, beta, out);
  k_out<<<(N_ * OUT_ + 255) / 256, 256, 0, stream>>>(out, Wo, bo, out + (size_t)N_ * H_);
}
